// EdgeDecoder_10471130268373
// MI455X (gfx1250) — matrix operand order repaired and checked by reading, not compiled, not run on hardware
//
#include <hip/hip_runtime.h>
#include <hip/hip_bf16.h>

#define HID   128
#define K2    256          // 2*HID concat width
#define K2P   264          // padded bf16 row length (528B -> bank-conflict-free b128 reads)
#define EPW   16           // edges per wave (M tile)
#define WPW   8            // waves per workgroup
#define EPWG  (EPW * WPW)  // 128 edges per workgroup

typedef __attribute__((ext_vector_type(16))) __bf16 v16bf;
typedef __attribute__((ext_vector_type(8)))  __bf16 v8bf;
typedef __attribute__((ext_vector_type(4)))  __bf16 v4bf;
typedef __attribute__((ext_vector_type(8)))  float  v8f;

__global__ __launch_bounds__(256, 1)
void edge_decoder_wmma(const float* __restrict__ z,
                       const long long* __restrict__ eli,
                       const float* __restrict__ W1,
                       const float* __restrict__ b1,
                       const float* __restrict__ W2,
                       const float* __restrict__ b2,
                       float* __restrict__ out,
                       long long E)
{
    __shared__ __bf16 sW1t[HID * K2P];      // W1 transposed [n][k], bf16
    __shared__ __bf16 sA[WPW][EPW * K2P];   // gathered+concat A tiles, per wave

    const int tid  = threadIdx.x;
    const int lane = tid & 31;
    const int wave = tid >> 5;
    const int half = lane >> 4;             // 0: lanes 0-15, 1: lanes 16-31
    const int mrow = lane & 15;

    // ---- Stage W1 (k-major [256][128] f32) into LDS transposed bf16 ----
    for (int i = tid; i < K2 * HID; i += 256) {
        int k = i >> 7;             // / HID
        int n = i & (HID - 1);
        sW1t[n * K2P + k] = (__bf16)W1[i];
    }
    __syncthreads();

    const long long e0 = (long long)blockIdx.x * EPWG + (long long)wave * EPW;

    // ---- Gather 16 edges: row-node (K 0..127) and col-node (K 128..255) ----
    __bf16* aw = sA[wave];
    for (int r = 0; r < EPW; ++r) {
        long long e = e0 + r; if (e >= E) e = E - 1;
        const long long nr = eli[e];        // edge_label_index[0][e]
        const long long nc = eli[E + e];    // edge_label_index[1][e]
        const float4 fr = *(const float4*)(z + nr * HID + lane * 4);
        const float4 fc = *(const float4*)(z + nc * HID + lane * 4);
        v4bf vr, vc;
        vr[0] = (__bf16)fr.x; vr[1] = (__bf16)fr.y;
        vr[2] = (__bf16)fr.z; vr[3] = (__bf16)fr.w;
        vc[0] = (__bf16)fc.x; vc[1] = (__bf16)fc.y;
        vc[2] = (__bf16)fc.z; vc[3] = (__bf16)fc.w;
        *(v4bf*)(aw + r * K2P +        lane * 4) = vr;
        *(v4bf*)(aw + r * K2P + HID +  lane * 4) = vc;
    }
    // (wave-local LDS producer/consumer; compiler inserts s_wait_dscnt)

    // ---- A fragments: 16x256 bf16 = 8 K-blocks of 16x32 ----
    // Layout: lane<16 -> M=lane, K = kb*32 + {0..7, 16..23}
    //         lane>=16 -> M=lane-16, K = kb*32 + {8..15, 24..31}
    v16bf afrag[8];
    for (int kb = 0; kb < 8; ++kb) {
        const __bf16* p = aw + mrow * K2P + kb * 32 + half * 8;
        v8bf lo = *(const v8bf*)(p);
        v8bf hi = *(const v8bf*)(p + 16);
        afrag[kb] = __builtin_shufflevector(lo, hi,
                        0,1,2,3,4,5,6,7, 8,9,10,11,12,13,14,15);
    }

    const float b2v = b2[0];
    float acc[8];
    for (int r = 0; r < 8; ++r) acc[r] = 0.0f;

    // ---- N tiles: h = relu(A*W1 + b1); acc += h * W2 ----
    for (int t = 0; t < 8; ++t) {
        const int n = mrow + t * 16;
        const __bf16* bp = sW1t + n * K2P;
        v8f c = {};
        for (int kb = 0; kb < 8; ++kb) {
            // B layout: lanes 0-15 hold K kb*32+{0..7,16..23}, lanes 16-31 hold kb*32+{8..15,24..31}
            const __bf16* q = bp + kb * 32 + half * 8;
            v8bf lo = *(const v8bf*)(q);
            v8bf hi = *(const v8bf*)(q + 16);
            v16bf bfrag = __builtin_shufflevector(lo, hi,
                              0,1,2,3,4,5,6,7, 8,9,10,11,12,13,14,15);
            c = __builtin_amdgcn_wmma_f32_16x16x32_bf16(
                    /*neg_a=*/false, afrag[kb], /*neg_b=*/false, bfrag,
                    /*c_mod=*/(short)0, c, /*reuse_a=*/false, /*reuse_b=*/false);
        }
        const float b1n = b1[n];
        const float w2n = W2[n];
        for (int r = 0; r < 8; ++r) {
            float h = c[r] + b1n;
            h = h > 0.0f ? h : 0.0f;
            acc[r] += h * w2n;
        }
    }

    // ---- Reduce over the 16 lanes of each half (N columns) ----
    for (int off = 8; off >= 1; off >>= 1)
        for (int r = 0; r < 8; ++r)
            acc[r] += __shfl_xor(acc[r], off, 32);

    // D layout: lanes 0-15 hold M=0..7 (vgpr r), lanes 16-31 hold M=8..15
    if ((lane & 15) == 0) {
        const long long ebase = e0 + half * 8;
        for (int r = 0; r < 8; ++r) {
            long long e = ebase + r;
            if (e < E) out[e] = acc[r] + b2v;
        }
    }
}

extern "C" void kernel_launch(void* const* d_in, const int* in_sizes, int n_in,
                              void* d_out, int out_size, void* d_ws, size_t ws_size,
                              hipStream_t stream) {
    const float*     z   = (const float*)d_in[0];
    const long long* eli = (const long long*)d_in[1];   // int64 per reference
    const float*     W1  = (const float*)d_in[2];
    const float*     b1  = (const float*)d_in[3];
    const float*     W2  = (const float*)d_in[4];
    const float*     b2  = (const float*)d_in[5];
    float* out = (float*)d_out;

    const long long E = (long long)in_sizes[1] / 2;
    const int nwg = (int)((E + EPWG - 1) / EPWG);
    if (nwg > 0) {
        edge_decoder_wmma<<<nwg, 256, 0, stream>>>(z, eli, W1, b1, W2, b2, out, E);
    }
}
